// GAT_56719338111588
// MI455X (gfx1250) — compile-verified, matrix operand order checked by reading, not run on hardware
//
#include <hip/hip_runtime.h>
#include <hip/hip_bf16.h>

typedef __attribute__((ext_vector_type(16))) _Float16 v16h;
typedef __attribute__((ext_vector_type(8)))  _Float16 v8h;
typedef __attribute__((ext_vector_type(8)))  float    v8f;
typedef int v4i_ls __attribute__((vector_size(16)));   // matches async-LDS builtin pointee

#define N_NODES 20000
#define N_EDGES 320000
#define BGRAPH  8
#define HDIM    256      // HEADS * {HID,OUT} = 4*64 for every layer

#define AS1 __attribute__((address_space(1)))
#define AS3 __attribute__((address_space(3)))

// ---------------- small helpers ----------------
__device__ __forceinline__ unsigned f2ord(float f) {
  unsigned b = __float_as_uint(f);
  return (b & 0x80000000u) ? ~b : (b | 0x80000000u);
}
__device__ __forceinline__ float ord2f(unsigned u) {
  unsigned b = (u & 0x80000000u) ? (u ^ 0x80000000u) : ~u;
  return __uint_as_float(b);
}
__device__ __forceinline__ float elu1(float x) {
  return x > 0.f ? x : (__expf(x) - 1.f);
}

__global__ void k_fill_f32(float* p, float v, size_t n) {
  size_t i = (size_t)blockIdx.x * blockDim.x + threadIdx.x;
  if (i < n) p[i] = v;
}
__global__ void k_fill_u32(unsigned* p, unsigned v, size_t n) {
  size_t i = (size_t)blockIdx.x * blockDim.x + threadIdx.x;
  if (i < n) p[i] = v;
}
__global__ void k_cast16(const float* __restrict__ a, _Float16* __restrict__ b, size_t n) {
  size_t i = (size_t)blockIdx.x * blockDim.x + threadIdx.x;
  if (i < n) b[i] = (_Float16)a[i];
}
// W (K x 256) f32 -> Wt (256 x K) f16
__global__ void k_transpose(const float* __restrict__ W, _Float16* __restrict__ Wt, int K) {
  int i = blockIdx.x * blockDim.x + threadIdx.x;
  if (i >= K * 256) return;
  int k = i >> 8, m = i & 255;
  Wt[(size_t)m * K + k] = (_Float16)W[i];
}

// h_aa = emb[ref_aa] * emb[alt_aa[gid]] ; h0 = concat(h_aa, node_feat) -> f16 (N x 128)
__global__ void k_embed(const float* __restrict__ node_feat, const int* __restrict__ ref_aa,
                        const int* __restrict__ alt_aa, const int* __restrict__ gid,
                        const float* __restrict__ emb, _Float16* __restrict__ h16) {
  int i = blockIdx.x * blockDim.x + threadIdx.x;
  if (i >= N_NODES * 128) return;
  int n = i >> 7, c = i & 127;
  float v;
  if (c < 64) v = emb[ref_aa[n] * 64 + c] * emb[alt_aa[gid[n]] * 64 + c];
  else        v = node_feat[(size_t)n * 64 + (c - 64)];
  h16[i] = (_Float16)v;
}

// ---------------- WMMA GEMM ----------------
// C(M x 256) = A(M x K, f16 row-major) * Bt(256 x K, f16 row-major)^T (+bias)
// grid = (M/16), block = 256 = 8 waves. One block covers 16 rows x 256 cols so A is
// streamed from HBM exactly once. Wave w owns column tiles {2w, 2w+1} (32 cols) with
// two f32 accumulators, reusing the A fragment from registers.
// A strip is staged to LDS via GLOBAL_LOAD_ASYNC_TO_LDS_B128 (ASYNCcnt) when available.
__global__ __launch_bounds__(256)
void k_wmma_gemm(const _Float16* __restrict__ A, const _Float16* __restrict__ Bt,
                 const float* __restrict__ bias, float* __restrict__ C,
                 int M, int K, int Nout) {
  __shared__ _Float16 As[16 * 264];           // K <= 256, ldk = K+8 keeps 16B alignment
  const int ldk  = K + 8;
  const int row0 = blockIdx.x * 16;
  const int tid  = threadIdx.x;

  // ---- stage 16 x K A strip into LDS (16B chunks) ----
  const int cpr = K >> 3;                     // 16-byte chunks per row
  for (int i = tid; i < 16 * cpr; i += 256) {
    int r = i / cpr, c = i - r * cpr;
    const _Float16* g = A + (size_t)(row0 + r) * K + c * 8;
    _Float16*       l = &As[r * ldk + c * 8];
#if __has_builtin(__builtin_amdgcn_global_load_async_to_lds_b128)
    __builtin_amdgcn_global_load_async_to_lds_b128(
        (AS1 v4i_ls*)(AS1 void*)g, (AS3 v4i_ls*)(AS3 void*)l, 0, 0);
#else
    *(v8h*)l = *(const v8h*)g;
#endif
  }
#if __has_builtin(__builtin_amdgcn_global_load_async_to_lds_b128)
#if __has_builtin(__builtin_amdgcn_s_wait_asynccnt)
  __builtin_amdgcn_s_wait_asynccnt(0);
#else
  asm volatile("s_wait_asynccnt 0x0" ::: "memory");
#endif
#endif
  __syncthreads();

  const int wave  = tid >> 5;
  const int lane  = tid & 31;
  const int colt0 = wave * 32;                // two 16-col tiles per wave
  const int rl    = lane & 15;
  const int kh    = lane >> 4;                // K-half select per ISA A/B layout

  v8f acc0 = {}, acc1 = {};
  const _Float16* bp0 = Bt + (size_t)(colt0 + rl) * K;
  const _Float16* bp1 = Bt + (size_t)(colt0 + 16 + rl) * K;

  for (int k0 = 0; k0 < K; k0 += 32) {
    v8h a0 = *(const v8h*)&As[rl * ldk + k0 + kh * 8];
    v8h a1 = *(const v8h*)&As[rl * ldk + k0 + 16 + kh * 8];
    v16h a = __builtin_shufflevector(a0, a1, 0,1,2,3,4,5,6,7,8,9,10,11,12,13,14,15);
    v8h b00 = *(const v8h*)(bp0 + k0 + kh * 8);
    v8h b01 = *(const v8h*)(bp0 + k0 + 16 + kh * 8);
    v16h b0 = __builtin_shufflevector(b00, b01, 0,1,2,3,4,5,6,7,8,9,10,11,12,13,14,15);
    v8h b10 = *(const v8h*)(bp1 + k0 + kh * 8);
    v8h b11 = *(const v8h*)(bp1 + k0 + 16 + kh * 8);
    v16h b1 = __builtin_shufflevector(b10, b11, 0,1,2,3,4,5,6,7,8,9,10,11,12,13,14,15);
    acc0 = __builtin_amdgcn_wmma_f32_16x16x32_f16(false, a, false, b0, (short)0, acc0, false, false);
    acc1 = __builtin_amdgcn_wmma_f32_16x16x32_f16(false, a, false, b1, (short)0, acc1, false, false);
  }

  float bv0 = bias ? bias[colt0 + rl]      : 0.f;
  float bv1 = bias ? bias[colt0 + 16 + rl] : 0.f;
#pragma unroll
  for (int r = 0; r < 8; ++r) {
    int m = r + kh * 8;                      // C/D layout: VGPR r -> M=r (lanes 0-15), M=r+8 (16-31)
    C[(size_t)(row0 + m) * Nout + colt0 + rl]      = acc0[r] + bv0;
    C[(size_t)(row0 + m) * Nout + colt0 + 16 + rl] = acc1[r] + bv1;
  }
}

// ---------------- edge kernels ----------------
// one wave per edge: f_out = leaky(ni[src]+nj[dst]+fe); elog[h] = sum_d f_out*attn
__global__ __launch_bounds__(256)
void k_edge_attn(const float* __restrict__ ni, const float* __restrict__ nj,
                 const float* __restrict__ fe, const int* __restrict__ src,
                 const int* __restrict__ dst, const float* __restrict__ attn,
                 float* __restrict__ fout, float* __restrict__ elog) {
  int e = (blockIdx.x * blockDim.x + threadIdx.x) >> 5;
  int lane = threadIdx.x & 31;
  if (e >= N_EDGES) return;
  const float* pni = ni + (size_t)src[e] * HDIM;
  const float* pnj = nj + (size_t)dst[e] * HDIM;
  const float* pfe = fe + (size_t)e * HDIM;
  float*       po  = fout + (size_t)e * HDIM;
  float partial = 0.f;
  int base = lane * 8;
#pragma unroll
  for (int j = 0; j < 8; ++j) {
    int idx = base + j;
    float v = pni[idx] + pnj[idx] + pfe[idx];
    v = v > 0.f ? v : 0.01f * v;             // leaky_relu(0.01)
    po[idx] = v;
    partial += v * attn[idx];                // attn flat (h*64+d) matches idx
  }
  partial += __shfl_xor(partial, 4, 8);      // reduce within 8-lane head group
  partial += __shfl_xor(partial, 2, 8);
  partial += __shfl_xor(partial, 1, 8);
  if ((lane & 7) == 0) elog[(size_t)e * 4 + (lane >> 3)] = partial;
}

__global__ void k_segmax(const float* __restrict__ elog, const int* __restrict__ dst,
                         unsigned* __restrict__ menc) {
  int i = blockIdx.x * blockDim.x + threadIdx.x;
  if (i >= N_EDGES * 4) return;
  int e = i >> 2, h = i & 3;
  atomicMax(&menc[dst[e] * 4 + h], f2ord(elog[i]));
}

__global__ void k_softexp(const float* __restrict__ elog, const int* __restrict__ dst,
                          const unsigned* __restrict__ menc, float* __restrict__ ex,
                          float* __restrict__ ssum) {
  int i = blockIdx.x * blockDim.x + threadIdx.x;
  if (i >= N_EDGES * 4) return;
  int e = i >> 2, h = i & 3;
  float m = ord2f(menc[dst[e] * 4 + h]);
  float v = __expf(elog[i] - m);
  ex[i] = v;
  atomicAdd(&ssum[dst[e] * 4 + h], v);
}

// one wave per edge: hnew[dst] += hout[src] * a   (a per head)
__global__ __launch_bounds__(256)
void k_aggregate(const float* __restrict__ ex, const float* __restrict__ ssum,
                 const int* __restrict__ src, const int* __restrict__ dst,
                 const float* __restrict__ hout, float* __restrict__ hnew) {
  int e = (blockIdx.x * blockDim.x + threadIdx.x) >> 5;
  int lane = threadIdx.x & 31;
  if (e >= N_EDGES) return;
  int d = dst[e], s = src[e], h = lane >> 3;
  float a = ex[(size_t)e * 4 + h] / ssum[d * 4 + h];
  const float* ph = hout + (size_t)s * HDIM;
  float*       pd = hnew + (size_t)d * HDIM;
  int base = lane * 8;
#pragma unroll
  for (int j = 0; j < 8; ++j) atomicAdd(&pd[base + j], ph[base + j] * a);
}

__global__ void k_elu_to_f16(const float* __restrict__ in, _Float16* __restrict__ out, size_t n) {
  size_t i = (size_t)blockIdx.x * blockDim.x + threadIdx.x;
  if (i < n) out[i] = (_Float16)elu1(in[i]);
}

// final layer: mean over 4 heads, elu, atomic mean-readout per graph
__global__ void k_head_mean_readout(const float* __restrict__ hnew, const int* __restrict__ gid,
                                    float* __restrict__ hg) {
  int i = blockIdx.x * blockDim.x + threadIdx.x;
  if (i >= N_NODES * 64) return;
  int n = i >> 6, d = i & 63;
  const float* p = hnew + (size_t)n * HDIM;
  float v = 0.25f * (p[d] + p[64 + d] + p[128 + d] + p[192 + d]);
  atomicAdd(&hg[gid[n] * 64 + d], elu1(v));
}

__global__ __launch_bounds__(256)
void k_mlp(const float* __restrict__ hg,
           const float* __restrict__ W0, const float* __restrict__ b0,
           const float* __restrict__ W1, const float* __restrict__ b1,
           const float* __restrict__ W2, const float* __restrict__ b2,
           float* __restrict__ out) {
  __shared__ float x[8 * 64], y0[8 * 32], y1[8 * 16];
  int t = threadIdx.x;
  for (int i = t; i < 512; i += 256) x[i] = hg[i] * (1.f / 2500.f);  // mean over 2500 nodes
  __syncthreads();
  { int g = t >> 5, j = t & 31;
    float s = b0[j];
    for (int k = 0; k < 64; ++k) s += x[g * 64 + k] * W0[k * 32 + j];
    y0[g * 32 + j] = fmaxf(s, 0.f); }
  __syncthreads();
  if (t < 128) {
    int g = t >> 4, j = t & 15;
    float s = b1[j];
    for (int k = 0; k < 32; ++k) s += y0[g * 32 + k] * W1[k * 16 + j];
    y1[g * 16 + j] = fmaxf(s, 0.f);
  }
  __syncthreads();
  if (t < 8) {
    float s = b2[0];
    for (int k = 0; k < 16; ++k) s += y1[t * 16 + k] * W2[k];
    out[t] = s;
  }
}

// ---------------- launch ----------------
extern "C" void kernel_launch(void* const* d_in, const int* in_sizes, int n_in,
                              void* d_out, int out_size, void* d_ws, size_t ws_size,
                              hipStream_t stream) {
  (void)in_sizes; (void)n_in; (void)out_size; (void)ws_size;
  const float* node_feat = (const float*)d_in[0];
  const float* edge_feat = (const float*)d_in[1];
  const int*   ref_aa    = (const int*)d_in[2];
  const int*   alt_aa    = (const int*)d_in[3];
  const int*   src       = (const int*)d_in[4];
  const int*   dst       = (const int*)d_in[5];
  const int*   gid       = (const int*)d_in[6];
  const float* emb       = (const float*)d_in[7];

  char* w = (char*)d_ws;
  auto alloc = [&](size_t bytes) -> void* {
    void* p = (void*)w; w += (bytes + 255) & ~(size_t)255; return p;
  };
  _Float16* h16  = (_Float16*)alloc((size_t)N_NODES * HDIM * 2);
  _Float16* f16b = (_Float16*)alloc((size_t)N_EDGES * HDIM * 2);
  float* ni   = (float*)alloc((size_t)N_NODES * HDIM * 4);
  float* nj   = (float*)alloc((size_t)N_NODES * HDIM * 4);
  float* hout = (float*)alloc((size_t)N_NODES * HDIM * 4);
  float* hnew = (float*)alloc((size_t)N_NODES * HDIM * 4);
  float* fe   = (float*)alloc((size_t)N_EDGES * HDIM * 4);
  float* fout = (float*)alloc((size_t)N_EDGES * HDIM * 4);
  float* elog = (float*)alloc((size_t)N_EDGES * 4 * 4);
  float* exs  = (float*)alloc((size_t)N_EDGES * 4 * 4);
  unsigned* menc = (unsigned*)alloc((size_t)N_NODES * 4 * 4);
  float* ssum = (float*)alloc((size_t)N_NODES * 4 * 4);
  _Float16* wtN  = (_Float16*)alloc(256 * 256 * 2);
  _Float16* wtNi = (_Float16*)alloc(256 * 256 * 2);
  _Float16* wtNj = (_Float16*)alloc(256 * 256 * 2);
  _Float16* wtF  = (_Float16*)alloc(256 * 256 * 2);
  float* hg = (float*)alloc(512 * 4);

  const int NB = 256;
  k_embed<<<(N_NODES * 128 + NB - 1) / NB, NB, 0, stream>>>(node_feat, ref_aa, alt_aa, gid, emb, h16);
  k_cast16<<<(N_EDGES * 32 + NB - 1) / NB, NB, 0, stream>>>(edge_feat, f16b, (size_t)N_EDGES * 32);

  for (int l = 0; l < 3; ++l) {
    int base = 8 + 6 * l;
    const float* Wn   = (const float*)d_in[base + 0];
    const float* bn   = (const float*)d_in[base + 1];
    const float* Wni  = (const float*)d_in[base + 2];
    const float* Wnj  = (const float*)d_in[base + 3];
    const float* Wf   = (const float*)d_in[base + 4];
    const float* attn = (const float*)d_in[base + 5];
    int Kh = (l == 0) ? 128 : 256;
    int Kf = (l == 0) ? 32 : 256;

    k_transpose<<<(Kh * 256 + NB - 1) / NB, NB, 0, stream>>>(Wn,  wtN,  Kh);
    k_transpose<<<(Kh * 256 + NB - 1) / NB, NB, 0, stream>>>(Wni, wtNi, Kh);
    k_transpose<<<(Kh * 256 + NB - 1) / NB, NB, 0, stream>>>(Wnj, wtNj, Kh);
    k_transpose<<<(Kf * 256 + NB - 1) / NB, NB, 0, stream>>>(Wf,  wtF,  Kf);

    dim3 gN(N_NODES / 16), gE(N_EDGES / 16);
    k_wmma_gemm<<<gN, NB, 0, stream>>>(h16,  wtNi, nullptr, ni,   N_NODES, Kh, HDIM);
    k_wmma_gemm<<<gN, NB, 0, stream>>>(h16,  wtNj, nullptr, nj,   N_NODES, Kh, HDIM);
    k_wmma_gemm<<<gN, NB, 0, stream>>>(h16,  wtN,  bn,      hout, N_NODES, Kh, HDIM);
    k_wmma_gemm<<<gE, NB, 0, stream>>>(f16b, wtF,  nullptr, fe,   N_EDGES, Kf, HDIM);

    k_edge_attn<<<N_EDGES / 8, NB, 0, stream>>>(ni, nj, fe, src, dst, attn, fout, elog);

    k_fill_u32<<<(N_NODES * 4 + NB - 1) / NB, NB, 0, stream>>>(menc, 0x007FFFFFu, (size_t)N_NODES * 4);
    k_fill_f32<<<(N_NODES * 4 + NB - 1) / NB, NB, 0, stream>>>(ssum, 0.f, (size_t)N_NODES * 4);
    k_fill_f32<<<((size_t)N_NODES * HDIM + NB - 1) / NB, NB, 0, stream>>>(hnew, 0.f, (size_t)N_NODES * HDIM);

    k_segmax<<<(N_EDGES * 4) / NB, NB, 0, stream>>>(elog, dst, menc);
    k_softexp<<<(N_EDGES * 4) / NB, NB, 0, stream>>>(elog, dst, menc, exs, ssum);
    k_aggregate<<<N_EDGES / 8, NB, 0, stream>>>(exs, ssum, src, dst, hout, hnew);

    if (l < 2) {
      k_elu_to_f16<<<((size_t)N_NODES * HDIM + NB - 1) / NB, NB, 0, stream>>>(hnew, h16, (size_t)N_NODES * HDIM);
      k_elu_to_f16<<<((size_t)N_EDGES * HDIM + NB - 1) / NB, NB, 0, stream>>>(fout, f16b, (size_t)N_EDGES * HDIM);
    }
  }

  k_fill_f32<<<2, NB, 0, stream>>>(hg, 0.f, 512);
  k_head_mean_readout<<<(N_NODES * 64) / NB, NB, 0, stream>>>(hnew, gid, hg);
  k_mlp<<<1, NB, 0, stream>>>(hg, (const float*)d_in[26], (const float*)d_in[27],
                              (const float*)d_in[28], (const float*)d_in[29],
                              (const float*)d_in[30], (const float*)d_in[31], (float*)d_out);
}
